// MultiTokenPred_34291018891302
// MI455X (gfx1250) — matrix operand order repaired and checked by reading, not compiled, not run on hardware
//
#include <hip/hip_runtime.h>
#include <hip/hip_bf16.h>

// ---------------- problem constants (from reference) ----------------
#define B_   4
#define T_   256
#define D_   1024
#define V_   32000
#define K_   4
#define H_   16
#define HD_  64          // D/H
#define N_   252         // T - TOT
#define M_   (B_*N_)     // 1008 rows per GEMM

typedef __attribute__((ext_vector_type(16))) __bf16 v16bf;
typedef __attribute__((ext_vector_type(8)))  __bf16 v8bf;
typedef __attribute__((ext_vector_type(8)))  float  v8f;

// =====================================================================
// fp32 -> bf16 bulk conversion (weights, once per launch)
// =====================================================================
__global__ void cvt_f32_bf16_kernel(const float* __restrict__ in,
                                    __bf16* __restrict__ out, long n) {
  long i = (long)blockIdx.x * blockDim.x + threadIdx.x;
  long stride = (long)gridDim.x * blockDim.x;
  for (; i < n; i += stride) out[i] = (__bf16)in[i];
}

// =====================================================================
// fused rmsnorm(h) || rmsnorm(fut) -> merged bf16, row r = n*B + b
// hsrc indexed as hsrc[b*hs_b + n*hs_n] (handles both (B,T,D) and (n,b) layouts)
// =====================================================================
__global__ __launch_bounds__(256)
void rmsnorm_concat_kernel(const float* __restrict__ hsrc, long hs_b, long hs_n,
                           const float* __restrict__ emb, int k,
                           const float* __restrict__ rms_w,
                           __bf16* __restrict__ merged) {
  int r = blockIdx.x;            // r = n*B + b
  int n = r / B_, b = r % B_;
  const float* hrow = hsrc + (long)b * hs_b + (long)n * hs_n;
  const float* frow = emb + ((long)b * T_ + (n + k + 1)) * D_;
  __shared__ float s0[256], s1[256];
  int t = threadIdx.x;
  float a0 = 0.f, a1 = 0.f;
  for (int c = t; c < D_; c += 256) {
    float h = hrow[c]; a0 += h * h;
    float f = frow[c]; a1 += f * f;
  }
  s0[t] = a0; s1[t] = a1; __syncthreads();
  for (int off = 128; off > 0; off >>= 1) {
    if (t < off) { s0[t] += s0[t + off]; s1[t] += s1[t + off]; }
    __syncthreads();
  }
  float inv0 = rsqrtf(s0[0] * (1.0f / D_) + 1e-6f);
  float inv1 = rsqrtf(s1[0] * (1.0f / D_) + 1e-6f);
  __bf16* mrow = merged + (long)r * (2 * D_);
  for (int c = t; c < D_; c += 256) {
    mrow[c]       = (__bf16)(hrow[c] * inv0 * rms_w[c]);
    mrow[D_ + c]  = (__bf16)(frow[c] * inv1 * rms_w[c]);
  }
}

// =====================================================================
// row LayerNorm fp32 -> bf16
// =====================================================================
__global__ __launch_bounds__(256)
void layernorm_bf16_kernel(const float* __restrict__ x,
                           const float* __restrict__ w,
                           const float* __restrict__ bia,
                           __bf16* __restrict__ y, int Dd) {
  int r = blockIdx.x;
  const float* row = x + (long)r * Dd;
  __shared__ float s[256], s2[256];
  int t = threadIdx.x;
  float a = 0.f, q = 0.f;
  for (int c = t; c < Dd; c += 256) { float v = row[c]; a += v; q += v * v; }
  s[t] = a; s2[t] = q; __syncthreads();
  for (int off = 128; off > 0; off >>= 1) {
    if (t < off) { s[t] += s[t + off]; s2[t] += s2[t + off]; }
    __syncthreads();
  }
  float mean = s[0] / Dd;
  float var  = s2[0] / Dd - mean * mean;
  float inv  = rsqrtf(var + 1e-5f);
  __bf16* orow = y + (long)r * Dd;
  for (int c = t; c < Dd; c += 256)
    orow[c] = (__bf16)((row[c] - mean) * inv * w[c] + bia[c]);
}

// =====================================================================
// tiny causal attention over batch dim (B=4 keys), per (n, head)
// qkv rows r = n*B+b, 3*D cols: [q | k | v]. output bf16, rows n*B+b.
// =====================================================================
__global__ __launch_bounds__(256)
void attention_kernel(const float* __restrict__ qkv, __bf16* __restrict__ o) {
  int n = blockIdx.x;
  __shared__ float att[H_][B_][B_];
  int t = threadIdx.x;
  const float scale = 0.125f;      // 1/sqrt(64)
  if (t < H_ * B_) {
    int h = t >> 2, bq = t & 3;
    const float* qrow = qkv + ((long)(n * B_ + bq)) * 3 * D_ + h * HD_;
    float sc[B_];
    float mx = -1e30f;
    for (int bk = 0; bk < B_; ++bk) {
      if (bk <= bq) {
        const float* krow = qkv + ((long)(n * B_ + bk)) * 3 * D_ + D_ + h * HD_;
        float s = 0.f;
        for (int d = 0; d < HD_; ++d) s += qrow[d] * krow[d];
        sc[bk] = s * scale;
      } else sc[bk] = -1e9f;
      mx = fmaxf(mx, sc[bk]);
    }
    float sum = 0.f;
    for (int bk = 0; bk < B_; ++bk) { sc[bk] = __expf(sc[bk] - mx); sum += sc[bk]; }
    float is = 1.0f / sum;
    for (int bk = 0; bk < B_; ++bk) att[h][bq][bk] = sc[bk] * is;
  }
  __syncthreads();
  for (int idx = t; idx < B_ * D_; idx += 256) {
    int bq = idx / D_, c = idx % D_;
    int h = c / HD_;
    float acc = 0.f;
    for (int bk = 0; bk <= bq; ++bk)
      acc += att[h][bq][bk] * qkv[((long)(n * B_ + bk)) * 3 * D_ + 2 * D_ + c];
    o[((long)(n * B_ + bq)) * D_ + c] = (__bf16)acc;
  }
}

// =====================================================================
// reorder state (n,b) -> (b,n) + fp32->bf16 for LM head input
// =====================================================================
__global__ void reorder_bf16_kernel(const float* __restrict__ x,
                                    __bf16* __restrict__ hb) {
  long i = (long)blockIdx.x * blockDim.x + threadIdx.x;
  long tot = (long)M_ * D_;
  long stride = (long)gridDim.x * blockDim.x;
  for (; i < tot; i += stride) {
    long row = i / D_, c = i % D_;      // row = b*N + n
    long b = row / N_, n = row % N_;
    hb[i] = (__bf16)x[((long)n * B_ + b) * D_ + c];
  }
}

// =====================================================================
// bf16 WMMA GEMM: out[m,n] = sum_k A[m,k] * W[n,k] + bias[n] (+resid) (+gelu)
// A: M x Kd row-major bf16;  W: Nout x Kd row-major bf16 (weights are (out,in))
// wave32 tile: 32(M) x 64(N) = 8 chained v_wmma_f32_16x16x32_bf16 per 32-K step.
// block: 256 threads = 8 waves (2 M x 4 N) -> 64(M) x 256(N) block tile.
// K-loop: 8-stage unroll (K step 256) with ping-pong fragment buffers. Stage
// offsets are compile-time (fold into global_load immediate offsets), so VGPR
// pointer bumps happen once per 256-K — fewer v_add_nc_u64 ops, fewer
// WMMA->VALU hazard v_nops, and 12 B128 loads stay in flight behind 8 WMMAs.
// Stage 7 prefetches 64B past Kd on the last iteration: all GEMM operands
// live in workspace slabs padded by 1KB, so the over-read is harmless.
// =====================================================================
__device__ inline float gelu_tanh(float x) {
  const float c = 0.7978845608028654f;   // sqrt(2/pi)
  return 0.5f * x * (1.0f + tanhf(c * (x + 0.044715f * x * x * x)));
}

template <bool GELU, bool BF16OUT, bool NTSTORE>
__global__ __launch_bounds__(256)
void gemm_bf16_wmma(const __bf16* __restrict__ A, const __bf16* __restrict__ W,
                    const float* __restrict__ bias, const float* __restrict__ resid,
                    float* __restrict__ outf, __bf16* __restrict__ outb,
                    int M, int Nout, int Kd, long ldc) {
  const int lane = threadIdx.x & 31;
  const int wave = threadIdx.x >> 5;   // 0..7
  const int wm   = wave & 1;           // 2 waves along M
  const int wn   = wave >> 1;          // 4 waves along N
  const int half = lane >> 4;          // lane group 0/1
  const int r    = lane & 15;

  const int row0 = blockIdx.y * 64 + wm * 32;
  const int col0 = blockIdx.x * 256 + wn * 64;

  // clamped per-lane row pointers (stores are guarded in the epilogue)
  int ar0 = row0 + r;       if (ar0 >= M) ar0 = M - 1;
  int ar1 = row0 + 16 + r;  if (ar1 >= M) ar1 = M - 1;
  const __bf16* Ap[2] = { A + (long)ar0 * Kd, A + (long)ar1 * Kd };
  const __bf16* Wp[4];
#pragma unroll
  for (int t = 0; t < 4; ++t) {
    int wr = col0 + t * 16 + r; if (wr >= Nout) wr = Nout - 1;
    Wp[t] = W + (long)wr * Kd;
  }

  // A fragment per ISA 16-bit A 16x32 layout:
  //  lanes 0-15: elems0-7 = K[kk..+7],  elems8-15 = K[kk+16..+23]
  //  lanes16-31: elems0-7 = K[kk+8..+15], elems8-15 = K[kk+24..+31]
  auto loadA = [&](int s, int kk) -> v16bf {
    union { v16bf v; v8bf h[2]; } u;
    u.h[0] = *(const v8bf*)(Ap[s] + kk + half * 8);
    u.h[1] = *(const v8bf*)(Ap[s] + kk + 16 + half * 8);
    return u.v;
  };
  // B fragment: lane holds column (=weight row), same K order as A:
  // lanes 0-15: K[kk..+7],K[kk+16..+23]; lanes 16-31: K[kk+8..+15],K[kk+24..+31]
  auto loadB = [&](int t, int kk) -> v16bf {
    union { v16bf v; v8bf h[2]; } u;
    const __bf16* p = Wp[t] + kk + half * 8;
    u.h[0] = *(const v8bf*)(p);
    u.h[1] = *(const v8bf*)(p + 16);
    return u.v;
  };

  v8f acc[2][4] = {};

  v16bf abuf[2][2], bbuf[2][4];
#pragma unroll
  for (int s = 0; s < 2; ++s) abuf[0][s] = loadA(s, 0);
#pragma unroll
  for (int t = 0; t < 4; ++t) bbuf[0][t] = loadB(t, 0);

  for (int kk = 0; kk < Kd; kk += 256) {
#pragma unroll
    for (int st = 0; st < 8; ++st) {
      const int cur = st & 1, nxt = cur ^ 1;
      const int ko  = (st + 1) * 32;     // compile-time -> immediate offsets
      // issue next-stage fragment loads (12 B128s, in flight behind the WMMAs)
#pragma unroll
      for (int s = 0; s < 2; ++s) abuf[nxt][s] = loadA(s, kk + ko);
#pragma unroll
      for (int t = 0; t < 4; ++t) bbuf[nxt][t] = loadB(t, kk + ko);
      if (st == 0)
        __builtin_prefetch(Ap[0] + kk + 512, 0, 1);         // global_prefetch
      // compute on current stage's fragments
#pragma unroll
      for (int s = 0; s < 2; ++s)
#pragma unroll
        for (int t = 0; t < 4; ++t)
          acc[s][t] = __builtin_amdgcn_wmma_f32_16x16x32_bf16(
              false, abuf[cur][s], false, bbuf[cur][t], (short)0,
              acc[s][t], false, false);
    }
  }

  // epilogue: C/D layout -> m = row0 + s*16 + 8*half + j, n = col0 + t*16 + (lane&15)
#pragma unroll
  for (int s = 0; s < 2; ++s) {
#pragma unroll
    for (int t = 0; t < 4; ++t) {
      int n = col0 + t * 16 + r;
      if (n >= Nout) continue;
      float bv = bias[n];
#pragma unroll
      for (int j = 0; j < 8; ++j) {
        int m = row0 + s * 16 + half * 8 + j;
        if (m >= M) continue;
        float v = acc[s][t][j] + bv;
        if (resid) v += resid[(long)m * Nout + n];
        if (GELU)  v = gelu_tanh(v);
        if (BF16OUT) {
          outb[(long)m * ldc + n] = (__bf16)v;
        } else if (NTSTORE) {
          __builtin_nontemporal_store(v, outf + (long)m * ldc + n);
        } else {
          outf[(long)m * ldc + n] = v;
        }
      }
    }
  }
}

// =====================================================================
// host-side orchestration
// =====================================================================
extern "C" void kernel_launch(void* const* d_in, const int* in_sizes, int n_in,
                              void* d_out, int out_size, void* d_ws, size_t ws_size,
                              hipStream_t stream) {
  (void)in_sizes; (void)n_in; (void)out_size; (void)ws_size;

  const float* init_hidden = (const float*)d_in[0];   // (B,T,D)
  const float* token_emb   = (const float*)d_in[1];   // (B,T,D)
  const float* rms_w       = (const float*)d_in[2];   // (D)
  const float* lm_W        = (const float*)d_in[3];   // (V,D)
  const float* lm_b        = (const float*)d_in[4];   // (V)
  const float* proj_W      = (const float*)d_in[5];   // (K,D,2D)
  const float* proj_b      = (const float*)d_in[6];   // (K,D)
  const float* ln1_w       = (const float*)d_in[7];
  const float* ln1_b       = (const float*)d_in[8];
  const float* qkv_W       = (const float*)d_in[9];   // (K,3D,D)
  const float* qkv_b       = (const float*)d_in[10];  // (K,3D)
  const float* attn_o_W    = (const float*)d_in[11];  // (K,D,D)
  const float* attn_o_b    = (const float*)d_in[12];
  const float* ln2_w       = (const float*)d_in[13];
  const float* ln2_b       = (const float*)d_in[14];
  const float* mlp_W1      = (const float*)d_in[15];  // (K,4D,D)
  const float* mlp_b1      = (const float*)d_in[16];
  const float* mlp_W2      = (const float*)d_in[17];  // (K,D,4D)
  const float* mlp_b2      = (const float*)d_in[18];
  float* out = (float*)d_out;                         // (B,N,K,V) fp32

  // ---- carve workspace (256B aligned slabs, +1KB pad for K-loop lookahead) ----
  char* wp = (char*)d_ws;
  auto carve = [&](size_t bytes) -> char* {
    char* p = wp;
    wp += (bytes + 1024 + 255) & ~(size_t)255;
    return p;
  };
  __bf16* lmWb   = (__bf16*)carve((size_t)V_ * D_ * 2);
  __bf16* projWb = (__bf16*)carve((size_t)K_ * D_ * 2 * D_ * 2);
  __bf16* qkvWb  = (__bf16*)carve((size_t)K_ * 3 * D_ * D_ * 2);
  __bf16* aoWb   = (__bf16*)carve((size_t)K_ * D_ * D_ * 2);
  __bf16* m1Wb   = (__bf16*)carve((size_t)K_ * 4 * D_ * D_ * 2);
  __bf16* m2Wb   = (__bf16*)carve((size_t)K_ * D_ * 4 * D_ * 2);
  __bf16* merged = (__bf16*)carve((size_t)M_ * 2 * D_ * 2);
  float*  x      = (float*) carve((size_t)M_ * D_ * 4);
  float*  qkvf   = (float*) carve((size_t)M_ * 3 * D_ * 4);
  __bf16* ybf    = (__bf16*)carve((size_t)M_ * D_ * 2);
  __bf16* obf    = (__bf16*)carve((size_t)M_ * D_ * 2);
  __bf16* ffbf   = (__bf16*)carve((size_t)M_ * 4 * D_ * 2);
  __bf16* hb     = (__bf16*)carve((size_t)M_ * D_ * 2);

  // ---- one-time weight conversions fp32 -> bf16 ----
  cvt_f32_bf16_kernel<<<4096, 256, 0, stream>>>(lm_W,     lmWb,   (long)V_ * D_);
  cvt_f32_bf16_kernel<<<2048, 256, 0, stream>>>(proj_W,   projWb, (long)K_ * D_ * 2 * D_);
  cvt_f32_bf16_kernel<<<2048, 256, 0, stream>>>(qkv_W,    qkvWb,  (long)K_ * 3 * D_ * D_);
  cvt_f32_bf16_kernel<<<1024, 256, 0, stream>>>(attn_o_W, aoWb,   (long)K_ * D_ * D_);
  cvt_f32_bf16_kernel<<<2048, 256, 0, stream>>>(mlp_W1,   m1Wb,   (long)K_ * 4 * D_ * D_);
  cvt_f32_bf16_kernel<<<2048, 256, 0, stream>>>(mlp_W2,   m2Wb,   (long)K_ * D_ * 4 * D_);

  const int MB = (M_ + 63) / 64;                   // M tiles of 64
  dim3 gD   ((D_      + 255) / 256, MB);           // Nout = D
  dim3 gQKV ((3 * D_  + 255) / 256, MB);           // Nout = 3D
  dim3 gFF  ((4 * D_  + 255) / 256, MB);           // Nout = 4D
  dim3 gLM  ((V_      + 255) / 256, MB);           // Nout = V

  for (int k = 0; k < K_; ++k) {
    // merged = [rmsnorm(h) | rmsnorm(fut)]  (bf16, rows n*B+b)
    if (k == 0)
      rmsnorm_concat_kernel<<<M_, 256, 0, stream>>>(
          init_hidden, (long)T_ * D_, (long)D_, token_emb, k, rms_w, merged);
    else
      rmsnorm_concat_kernel<<<M_, 256, 0, stream>>>(
          x, (long)D_, (long)B_ * D_, token_emb, k, rms_w, merged);

    // x = merged @ proj_W[k]^T + proj_b[k]        (fp32, rows n*B+b)
    gemm_bf16_wmma<false, false, false><<<gD, 256, 0, stream>>>(
        merged, projWb + (long)k * D_ * 2 * D_, proj_b + (long)k * D_,
        nullptr, x, nullptr, M_, D_, 2 * D_, (long)D_);

    // y = LN1(x) -> bf16
    layernorm_bf16_kernel<<<M_, 256, 0, stream>>>(
        x, ln1_w + (long)k * D_, ln1_b + (long)k * D_, ybf, D_);

    // qkv = y @ qkv_W[k]^T + qkv_b[k]             (fp32)
    gemm_bf16_wmma<false, false, false><<<gQKV, 256, 0, stream>>>(
        ybf, qkvWb + (long)k * 3 * D_ * D_, qkv_b + (long)k * 3 * D_,
        nullptr, qkvf, nullptr, M_, 3 * D_, D_, (long)(3 * D_));

    // tiny causal attention over batch dim -> o (bf16)
    attention_kernel<<<N_, 256, 0, stream>>>(qkvf, obf);

    // x = x + o @ attn_o_W[k]^T + attn_o_b[k]     (fused residual, in place)
    gemm_bf16_wmma<false, false, false><<<gD, 256, 0, stream>>>(
        obf, aoWb + (long)k * D_ * D_, attn_o_b + (long)k * D_,
        x, x, nullptr, M_, D_, D_, (long)D_);

    // y2 = LN2(x) -> bf16
    layernorm_bf16_kernel<<<M_, 256, 0, stream>>>(
        x, ln2_w + (long)k * D_, ln2_b + (long)k * D_, ybf, D_);

    // ff = gelu(y2 @ mlp_W1[k]^T + b1)            (bf16, fused GELU epilogue)
    gemm_bf16_wmma<true, true, false><<<gFF, 256, 0, stream>>>(
        ybf, m1Wb + (long)k * 4 * D_ * D_, mlp_b1 + (long)k * 4 * D_,
        nullptr, nullptr, ffbf, M_, 4 * D_, D_, (long)(4 * D_));

    // x = x + ff @ mlp_W2[k]^T + b2               (fused residual, in place)
    gemm_bf16_wmma<false, false, false><<<gD, 256, 0, stream>>>(
        ffbf, m2Wb + (long)k * D_ * 4 * D_, mlp_b2 + (long)k * D_,
        x, x, nullptr, M_, D_, 4 * D_, (long)D_);

    // hb = bf16 state in (b,n) order for LM head
    reorder_bf16_kernel<<<1024, 256, 0, stream>>>(x, hb);

    // logits[:, :, k, :] = hb @ lm_W^T + lm_b
    // fp32, strided into (B,N,K,V); non-temporal stores keep lm_W hot in L2
    gemm_bf16_wmma<false, false, true><<<gLM, 256, 0, stream>>>(
        hb, lmWb, lm_b, nullptr, out + (long)k * V_, nullptr,
        M_, V_, D_, (long)K_ * V_);
  }
}
